// SparseAttention_55164559950476
// MI455X (gfx1250) — matrix operand order repaired and checked by reading, not compiled, not run on hardware
//
#include <hip/hip_runtime.h>

// ---------------- types ----------------
typedef unsigned short u16;
typedef unsigned int   u32;
typedef __attribute__((ext_vector_type(16))) __bf16 v16bf;
typedef __attribute__((ext_vector_type(8)))  float  v8f;
typedef __attribute__((ext_vector_type(4)))  u32    u32x4;
typedef __attribute__((ext_vector_type(2)))  u32    u32x2;
typedef __attribute__((ext_vector_type(4)))  float  f32x4;

// ---------------- problem constants ----------------
#define SEQ   4096
#define HDIM  1024
#define HEADS 16
#define HD    64
#define BS    128
#define NB    32
#define NRB   2
#define NG    3
#define NBE   4
#define MTOT  (NBE*SEQ)   // 16384

// ---------------- helpers ----------------
__device__ __forceinline__ u16 f2bf(float f){
    union { float f; u32 u; } x; x.f = f;
    u32 r = x.u + 0x7FFFu + ((x.u >> 16) & 1u);   // round-to-nearest-even
    return (u16)(r >> 16);
}

union FragU { u32x4 u[2]; v16bf v; };

// A-operand fragment (16x32 bf16): lane = row (lane&15); K = lhalf*8 + [0..7] and +16
__device__ __forceinline__ v16bf ldfragA(const u16* rowp, int lhalf){
    FragU f;
    f.u[0] = *(const u32x4*)(rowp + lhalf*8);
    f.u[1] = *(const u32x4*)(rowp + lhalf*8 + 16);
    return f.v;
}
// B-operand fragment (32x16 bf16, Bt row-major): lane = col; K = lhalf*8 + [0..7] and +16
__device__ __forceinline__ v16bf ldfragB(const u16* rowp, int lhalf){
    FragU f;
    f.u[0] = *(const u32x4*)(rowp + lhalf*8);
    f.u[1] = *(const u32x4*)(rowp + lhalf*8 + 16);
    return f.v;
}

__device__ __forceinline__ v8f wmma_bf16(v16bf a, v16bf b, v8f c){
    return __builtin_amdgcn_wmma_f32_16x16x32_bf16(false, a, false, b, (short)0, c, false, false);
}

// ---------------- weight fp32 -> bf16 ----------------
__global__ __launch_bounds__(256) void cvt_w(const float* __restrict__ src,
                                             u16* __restrict__ dst, int n4){
    int i = blockIdx.x * blockDim.x + threadIdx.x;
    if (i < n4){
        f32x4 f = ((const f32x4*)src)[i];
        union { u16 s[4]; u32x2 u; } o;
        o.s[0] = f2bf(f[0]); o.s[1] = f2bf(f[1]);
        o.s[2] = f2bf(f[2]); o.s[3] = f2bf(f[3]);
        ((u32x2*)dst)[i] = o.u;
    }
}

// ---------------- GEMM: C[M,N] = A[M,K] * W[N,K]^T + bias ----------------
// Double-buffered LDS (one barrier per K-step); global loads for tile t+2 are
// staged in registers a full iteration ahead, so their latency is hidden.
template<bool AF32, bool OUTB16>
__global__ __launch_bounds__(256, 2)
void gemm_nt(const void* __restrict__ Aptr, const u16* __restrict__ W,
             const float* __restrict__ bias, void* __restrict__ Cptr,
             int M, int N, int K)
{
    __shared__ u16 As[2][128][40];   // ping-pong 128 x 32, padded
    __shared__ u16 Bs[2][128][40];

    const int tid   = threadIdx.x;
    const int lane  = tid & 31;
    const int w     = tid >> 5;
    const int wm    = w >> 1;       // 0..3
    const int wn    = w & 1;        // 0..1
    const int lrow  = lane & 15;
    const int lhalf = lane >> 4;
    const int bm    = blockIdx.y * 128;
    const int bn    = blockIdx.x * 128;

    const int ldr  = tid >> 1;          // 0..127 row within tile
    const int kseg = (tid & 1) * 16;    // 0 or 16

    const float* Af = (const float*)Aptr;
    const u16*   Ah = (const u16*)Aptr;

    f32x4 sa_f[4];   // staged A (fp32 path)
    u32x4 sa_h[2];   // staged A (bf16 path)
    u32x4 sb[2];     // staged B

    auto loadStage = [&](int kk){
        if constexpr (AF32){
            const float* p = Af + (size_t)(bm + ldr) * K + kk + kseg;
            sa_f[0] = ((const f32x4*)p)[0];
            sa_f[1] = ((const f32x4*)p)[1];
            sa_f[2] = ((const f32x4*)p)[2];
            sa_f[3] = ((const f32x4*)p)[3];
        } else {
            const u16* p = Ah + (size_t)(bm + ldr) * K + kk + kseg;
            sa_h[0] = ((const u32x4*)p)[0];
            sa_h[1] = ((const u32x4*)p)[1];
        }
        const u16* q = W + (size_t)(bn + ldr) * K + kk + kseg;
        sb[0] = ((const u32x4*)q)[0];
        sb[1] = ((const u32x4*)q)[1];
    };
    auto storeStage = [&](int b){
        union { u32x4 u[2]; u16 s[16]; } pk;
        if constexpr (AF32){
            #pragma unroll
            for (int t = 0; t < 4; t++)
                #pragma unroll
                for (int e = 0; e < 4; e++)
                    pk.s[t*4 + e] = f2bf(sa_f[t][e]);
        } else {
            pk.u[0] = sa_h[0]; pk.u[1] = sa_h[1];
        }
        *(u32x4*)&As[b][ldr][kseg]     = pk.u[0];
        *(u32x4*)&As[b][ldr][kseg + 8] = pk.u[1];
        *(u32x4*)&Bs[b][ldr][kseg]     = sb[0];
        *(u32x4*)&Bs[b][ldr][kseg + 8] = sb[1];
    };

    v8f acc[2][4];
    #pragma unroll
    for (int i = 0; i < 2; i++)
        #pragma unroll
        for (int j = 0; j < 4; j++)
            #pragma unroll
            for (int v = 0; v < 8; v++) acc[i][j][v] = 0.f;

    const int T = K >> 5;           // number of 32-wide K tiles
    loadStage(0);
    storeStage(0);
    if (T > 1) loadStage(32);       // stage tile 1
    __syncthreads();

    for (int t = 0; t < T; t++){
        const int cur = t & 1;
        if (t + 1 < T) storeStage(cur ^ 1);       // commit staged tile t+1
        if (t + 2 < T) loadStage((t + 2) * 32);   // stage tile t+2 (latency hidden)

        v16bf afr[2], bfr[4];
        #pragma unroll
        for (int i = 0; i < 2; i++) afr[i] = ldfragA(&As[cur][wm*32 + i*16 + lrow][0], lhalf);
        #pragma unroll
        for (int j = 0; j < 4; j++) bfr[j] = ldfragB(&Bs[cur][wn*64 + j*16 + lrow][0], lhalf);
        __builtin_amdgcn_sched_barrier(0);
        #pragma unroll
        for (int i = 0; i < 2; i++)
            #pragma unroll
            for (int j = 0; j < 4; j++)
                acc[i][j] = wmma_bf16(afr[i], bfr[j], acc[i][j]);

        __syncthreads();
    }

    // ---- epilogue ----
    #pragma unroll
    for (int j = 0; j < 4; j++){
        int n = bn + wn*64 + j*16 + lrow;
        float bv = bias[n];
        #pragma unroll
        for (int i = 0; i < 2; i++){
            int mbase = bm + wm*32 + i*16 + 8*lhalf;
            #pragma unroll
            for (int v = 0; v < 8; v++){
                float val = acc[i][j][v] + bv;
                size_t idx = (size_t)(mbase + v) * N + n;
                if constexpr (OUTB16) ((u16*)Cptr)[idx] = f2bf(val);
                else                  ((float*)Cptr)[idx] = val;
            }
        }
    }
}

// ---------------- block-sparse flash attention ----------------
// grid: (NB, HEADS, NBE), block: 256 (8 waves; wave w owns query rows w*16..w*16+15)
__global__ __launch_bounds__(256, 2)
void attn_kernel(const u16* __restrict__ Q, const u16* __restrict__ Kbuf,
                 const u16* __restrict__ Vbuf, const int* __restrict__ rt,
                 u16* __restrict__ O)
{
    __shared__ u16 Ks[128][72];      // K block, row-major [kj][d]
    __shared__ u16 Vt[64][136];      // V block transposed [d][kj]
    __shared__ u16 Ps[8][16][136];   // per-wave P tile [m][kj]

    const int qb  = blockIdx.x;
    const int h   = blockIdx.y;
    const int be  = blockIdx.z;
    const int tid = threadIdx.x;
    const int lane  = tid & 31;
    const int w     = tid >> 5;
    const int lrow  = lane & 15;
    const int lhalf = lane >> 4;

    const size_t rowbase = (size_t)be * SEQ;
    const int hoff = h * HD;
    const float scale = 0.125f;   // 1/sqrt(64)

    // Q fragments stay in registers (16 rows x 64 d = 2 k-steps)
    v16bf qf[2];
    {
        const u16* qp = Q + (rowbase + (size_t)qb*BS + w*16 + lrow) * HDIM + hoff;
        qf[0] = ldfragA(qp,      lhalf);
        qf[1] = ldfragA(qp + 32, lhalf);
    }

    float mrow[8], lsum[8];
    v8f accO[4];
    #pragma unroll
    for (int v = 0; v < 8; v++){ mrow[v] = -3.0e38f; lsum[v] = 0.f; }
    #pragma unroll
    for (int jt = 0; jt < 4; jt++)
        #pragma unroll
        for (int v = 0; v < 8; v++) accO[jt][v] = 0.f;

    for (int g = 0; g < NG; g++){
        int blk = (g == 0) ? qb : rt[qb*NRB + (g-1)];
        __syncthreads();   // previous block's compute done before overwrite

        // load K block row-major: thread -> (kj = tid>>1, 32 d-elems)
        {
            int kj  = tid >> 1;
            int seg = (tid & 1) * 32;
            const u32x4* kp = (const u32x4*)(Kbuf + (rowbase + (size_t)blk*BS + kj) * HDIM + hoff + seg);
            u32x4 a = kp[0], b = kp[1], c = kp[2], d = kp[3];
            u32x4* dst = (u32x4*)&Ks[kj][seg];
            dst[0] = a; dst[1] = b; dst[2] = c; dst[3] = d;
        }
        // load V block transposed: thread -> (kj = tid&127, 32 d-elems)
        {
            int kj = tid & 127;
            int dh = tid >> 7;
            const u16* vp = Vbuf + (rowbase + (size_t)blk*BS + kj) * HDIM + hoff + dh*32;
            #pragma unroll
            for (int s = 0; s < 4; s++){
                union { u32x4 u; u16 e[8]; } vv;
                vv.u = *(const u32x4*)(vp + s*8);
                int db = dh*32 + s*8;
                #pragma unroll
                for (int i = 0; i < 8; i++) Vt[db + i][kj] = vv.e[i];
            }
        }
        // prefetch next gathered block while this one is consumed
        if (g + 1 < NG){
            int nblk = rt[qb*NRB + g];
            const u16* pk = Kbuf + (rowbase + (size_t)nblk*BS + (tid >> 1)) * HDIM + hoff + (tid & 1) * 32;
            __builtin_prefetch(pk, 0, 1);
            const u16* pv = Vbuf + (rowbase + (size_t)nblk*BS + (tid & 127)) * HDIM + hoff + (tid >> 7) * 32;
            __builtin_prefetch(pv, 0, 1);
        }
        __syncthreads();

        // S = Q * K^T  (16 x 128, fp32); batch fragment loads per k-step
        v8f accS[8];
        #pragma unroll
        for (int j = 0; j < 8; j++)
            #pragma unroll
            for (int v = 0; v < 8; v++) accS[j][v] = 0.f;
        #pragma unroll
        for (int ks = 0; ks < 2; ks++){
            v16bf bfr[8];
            #pragma unroll
            for (int j = 0; j < 8; j++) bfr[j] = ldfragB(&Ks[j*16 + lrow][ks*32], lhalf);
            __builtin_amdgcn_sched_barrier(0);
            #pragma unroll
            for (int j = 0; j < 8; j++) accS[j] = wmma_bf16(qf[ks], bfr[j], accS[j]);
        }

        // online softmax; row of vgpr v is m = 8*lhalf + v (C-layout)
        #pragma unroll
        for (int v = 0; v < 8; v++){
            float mx = accS[0][v];
            #pragma unroll
            for (int j = 1; j < 8; j++) mx = fmaxf(mx, accS[j][v]);
            #pragma unroll
            for (int off = 1; off < 16; off <<= 1) mx = fmaxf(mx, __shfl_xor(mx, off, 32));
            float mnew = fmaxf(mrow[v], mx * scale);
            float corr = __expf(mrow[v] - mnew);
            float psum = 0.f;
            #pragma unroll
            for (int j = 0; j < 8; j++){
                float p = __expf(accS[j][v]*scale - mnew);
                accS[j][v] = p;
                psum += p;
            }
            #pragma unroll
            for (int off = 1; off < 16; off <<= 1) psum += __shfl_xor(psum, off, 32);
            lsum[v] = lsum[v]*corr + psum;
            mrow[v] = mnew;
            #pragma unroll
            for (int jt = 0; jt < 4; jt++) accO[jt][v] *= corr;
        }

        // P (C-layout) -> LDS -> A-layout fragments
        #pragma unroll
        for (int j = 0; j < 8; j++)
            #pragma unroll
            for (int v = 0; v < 8; v++)
                Ps[w][8*lhalf + v][j*16 + lrow] = f2bf(accS[j][v]);

        v16bf pf[4];
        #pragma unroll
        for (int ks = 0; ks < 4; ks++) pf[ks] = ldfragA(&Ps[w][lrow][ks*32], lhalf);
        // O += P * V ; ks-outer so the 4 WMMAs per batch hit independent accumulators
        #pragma unroll
        for (int ks = 0; ks < 4; ks++){
            v16bf vfr[4];
            #pragma unroll
            for (int jt = 0; jt < 4; jt++) vfr[jt] = ldfragB(&Vt[jt*16 + lrow][ks*32], lhalf);
            __builtin_amdgcn_sched_barrier(0);
            #pragma unroll
            for (int jt = 0; jt < 4; jt++) accO[jt] = wmma_bf16(pf[ks], vfr[jt], accO[jt]);
        }
    }

    // normalize and emit bf16 for the output projection
    #pragma unroll
    for (int jt = 0; jt < 4; jt++){
        #pragma unroll
        for (int v = 0; v < 8; v++){
            float val = accO[jt][v] / lsum[v];
            size_t row = rowbase + (size_t)qb*BS + w*16 + 8*lhalf + v;
            O[row * HDIM + hoff + jt*16 + lrow] = f2bf(val);
        }
    }
}

// ---------------- host launcher ----------------
extern "C" void kernel_launch(void* const* d_in, const int* in_sizes, int n_in,
                              void* d_out, int out_size, void* d_ws, size_t ws_size,
                              hipStream_t stream)
{
    (void)in_sizes; (void)n_in; (void)out_size; (void)ws_size;
    const float* x  = (const float*)d_in[0];
    const float* wq = (const float*)d_in[1];
    const float* bq = (const float*)d_in[2];
    const float* wk = (const float*)d_in[3];
    const float* bk = (const float*)d_in[4];
    const float* wv = (const float*)d_in[5];
    const float* bv = (const float*)d_in[6];
    const float* wo = (const float*)d_in[7];
    const float* bo = (const float*)d_in[8];
    const int*   rt = (const int*)d_in[9];

    char* ws = (char*)d_ws;
    size_t off = 0;
    auto alloc = [&](size_t bytes) -> void* {
        void* p = ws + off;
        off += (bytes + 255) & ~(size_t)255;
        return p;
    };
    u16* wq_b  = (u16*)alloc((size_t)HDIM*HDIM*2);
    u16* wk_b  = (u16*)alloc((size_t)HDIM*HDIM*2);
    u16* wv_b  = (u16*)alloc((size_t)HDIM*HDIM*2);
    u16* wo_b  = (u16*)alloc((size_t)HDIM*HDIM*2);
    u16* Qbuf  = (u16*)alloc((size_t)MTOT*HDIM*2);
    u16* Kbuf  = (u16*)alloc((size_t)MTOT*HDIM*2);
    u16* Vbuf  = (u16*)alloc((size_t)MTOT*HDIM*2);
    u16* AObuf = (u16*)alloc((size_t)MTOT*HDIM*2);

    const int n4 = HDIM*HDIM/4;
    cvt_w<<<dim3((n4+255)/256), dim3(256), 0, stream>>>(wq, wq_b, n4);
    cvt_w<<<dim3((n4+255)/256), dim3(256), 0, stream>>>(wk, wk_b, n4);
    cvt_w<<<dim3((n4+255)/256), dim3(256), 0, stream>>>(wv, wv_b, n4);
    cvt_w<<<dim3((n4+255)/256), dim3(256), 0, stream>>>(wo, wo_b, n4);

    dim3 ggrid(HDIM/128, MTOT/128);   // (8, 128)
    gemm_nt<true,  true ><<<ggrid, dim3(256), 0, stream>>>(x, wq_b, bq, Qbuf, MTOT, HDIM, HDIM);
    gemm_nt<true,  true ><<<ggrid, dim3(256), 0, stream>>>(x, wk_b, bk, Kbuf, MTOT, HDIM, HDIM);
    gemm_nt<true,  true ><<<ggrid, dim3(256), 0, stream>>>(x, wv_b, bv, Vbuf, MTOT, HDIM, HDIM);

    attn_kernel<<<dim3(NB, HEADS, NBE), dim3(256), 0, stream>>>(Qbuf, Kbuf, Vbuf, rt, AObuf);

    gemm_nt<false, false><<<ggrid, dim3(256), 0, stream>>>(AObuf, wo_b, bo, d_out, MTOT, HDIM, HDIM);
}
